// BinaryTreeLSTM_7155415515513
// MI455X (gfx1250) — matrix operand order repaired and checked by reading, not compiled, not run on hardware
//
#include <hip/hip_runtime.h>
#include <hip/hip_bf16.h>
#include <math.h>

typedef __bf16 bf16_t;
typedef __attribute__((ext_vector_type(16))) __bf16 v16bf;
typedef __attribute__((ext_vector_type(8)))  __bf16 v8bf;
typedef __attribute__((ext_vector_type(8)))  float  v8f;

#define B_   64
#define L_   64
#define W_   512
#define H_   512
#define K2H  1024   // 2H
#define N5H  2560   // 5H
#define NT_TREE 512 // 16 wave32s per tree block (4 per SIMD32)

__device__ __forceinline__ float sigm(float x) { return 1.0f / (1.0f + __expf(-x)); }

// ---------------------------------------------------------------------------
// Kernel 0: f32 -> bf16 conversion of inp, Ww, Wc into workspace
// ---------------------------------------------------------------------------
__global__ void k_convert(const float* __restrict__ inp, const float* __restrict__ Ww,
                          const float* __restrict__ Wc,
                          bf16_t* __restrict__ inpB, bf16_t* __restrict__ WwB,
                          bf16_t* __restrict__ WcB) {
  const int NI = B_ * L_ * W_;   // 2097152
  const int NW = K2H * W_;       // 524288
  const int NC = N5H * K2H;      // 2621440
  const int total = NI + NW + NC;
  int stride = gridDim.x * blockDim.x;
  for (int idx = blockIdx.x * blockDim.x + threadIdx.x; idx < total; idx += stride) {
    if (idx < NI)            inpB[idx]           = (bf16_t)inp[idx];
    else if (idx < NI + NW)  WwB[idx - NI]       = (bf16_t)Ww[idx - NI];
    else                     WcB[idx - NI - NW]  = (bf16_t)Wc[idx - NI - NW];
  }
}

// ---------------------------------------------------------------------------
// Kernel 1: initial projection  st = inp @ Ww^T + bw  ->  h0 | c0  (f32 state)
//   M = B*L = 4096, K = W = 512, N = 2H = 1024.  One 16x16 tile per wave.
// ---------------------------------------------------------------------------
__global__ void __launch_bounds__(256) k_proj(const bf16_t* __restrict__ inpB,
                                              const bf16_t* __restrict__ WwB,
                                              const float* __restrict__ bw,
                                              float* __restrict__ hA,
                                              float* __restrict__ cA) {
  const int lane = threadIdx.x & 31;
  const int wave = threadIdx.x >> 5;
  const int lrow = lane & 15;      // tile row (A) / tile col (B,C)
  const int lhi  = lane >> 4;      // K-half select (A,B) / M+8 select (C)

  int tile = blockIdx.x * 8 + wave;        // 16384 tiles total
  int mt = tile >> 6;                      // 0..255
  int nt = tile & 63;                      // 0..63

  const int arow = mt * 16 + lrow;
  const int ncol = nt * 16 + lrow;
  const bf16_t* abase = inpB + (size_t)arow * W_;
  const bf16_t* bbase = WwB  + (size_t)ncol * W_;

  v8f acc = {};
  for (int kb = 0; kb < W_; kb += 32) {
    union { v16bf v; v8bf h2[2]; } au;
    const bf16_t* ap = abase + kb + 8 * lhi;
    au.h2[0] = *(const v8bf*)(ap);          // K = kb + 8*lhi     .. +7
    au.h2[1] = *(const v8bf*)(ap + 16);     // K = kb+16 + 8*lhi  .. +7
    union { v16bf v; v8bf h2[2]; } bu;
    bu.h2[0] = *(const v8bf*)(bbase + kb + 8 * lhi);        // K = kb + 8*lhi     .. +7
    bu.h2[1] = *(const v8bf*)(bbase + kb + 16 + 8 * lhi);   // K = kb+16 + 8*lhi  .. +7
    v16bf bfrag = bu.v;
    acc = __builtin_amdgcn_wmma_f32_16x16x32_bf16(false, au.v, false, bfrag,
                                                  (short)0, acc, false, false);
  }

  float bias = bw[ncol];
#pragma unroll
  for (int r = 0; r < 8; ++r) {
    int m  = mt * 16 + r + 8 * lhi;   // global row
    int bb = m >> 6;                  // batch
    int ll = m & 63;                  // position
    float val = acc[r] + bias;
    if (ncol < H_) hA[((size_t)bb * L_ + ll) * H_ + ncol]        = val;
    else           cA[((size_t)bb * L_ + ll) * H_ + (ncol - H_)] = val;
  }
}

// ---------------------------------------------------------------------------
// Kernel 2: persistent per-batch tree LSTM. One block per batch element.
//   Per iteration: stage h->LDS(bf16), WMMA GEMM (Lp x 1024)x(1024 x 2560)
//   fused with LSTM cell (5 gate accumulators, B-fragment reused across a
//   pair of m-tiles), scores+argmax, select/shift/mask, buffer swap.
// ---------------------------------------------------------------------------
__global__ void __launch_bounds__(NT_TREE) k_tree(const bf16_t* __restrict__ WcB,
                                                  const float* __restrict__ bc,
                                                  const float* __restrict__ q,
                                                  const int* __restrict__ length,
                                                  float* hA, float* cA,
                                                  float* hB, float* cB,
                                                  float* __restrict__ out) {
  __shared__ __align__(16) char smem[L_ * H_ * sizeof(bf16_t)];  // 64 KiB
  bf16_t* hlds = (bf16_t*)smem;           // staging buffer (phase 0/1)
  float*  part = (float*)smem;            // 512 partial dot sums (phase 2)
  float*  scf  = (float*)smem + NT_TREE;  // 64 scores
  int*    ksel = (int*)smem + NT_TREE + L_;

  const int b    = blockIdx.x;
  const int tid  = threadIdx.x;
  const int lane = tid & 31;
  const int wave = tid >> 5;              // 0..15
  const int lrow = lane & 15;
  const int lhi  = lane >> 4;
  const int len  = length[b];

  float* hc = hA + (size_t)b * L_ * H_;   // current state
  float* cc = cA + (size_t)b * L_ * H_;
  float* hn = hB + (size_t)b * L_ * H_;   // next state
  float* cn = cB + (size_t)b * L_ * H_;

  for (int i = 0; i < L_ - 1; ++i) {
    const int curLen = L_ - i;
    const int Lp     = curLen - 1;
    const bool active = (len > i);        // uniform across block
    int kstar = 0;

    if (active) {
      // ---- phase 0: stage h (f32 -> bf16) into LDS, zero-pad tail rows ----
      for (int idx = tid; idx < L_ * H_; idx += NT_TREE) {
        int row = idx >> 9;
        hlds[idx] = (row < curLen) ? (bf16_t)hc[idx] : (bf16_t)0.0f;
      }
      __syncthreads();

      // ---- phase 1: WMMA GEMM + fused LSTM cell -> hn/cn ----
      // combos: mp in [0,2) (pair of 16-row m-tiles), t in [0,32) (16-dim tile)
      // Each B fragment feeds 2 WMMAs (both m-tiles); A reused across 5 gates.
      for (int idx = wave; idx < 64; idx += 16) {
        int mp = idx >> 5;          // 0..1
        int t  = idx & 31;          // 0..31
        int M0 = mp * 32;
        if (M0 >= Lp) continue;                 // uniform per wave
        const bool doM1 = (M0 + 16) < Lp;       // second tile of the pair?
        v8f acc[2][5] = {};
        const int d = t * 16 + lrow;            // output dim / B column

        // warm L2/WGP$ for the 5 gate rows of this column (global_prefetch_b8)
#pragma unroll
        for (int g = 0; g < 5; ++g)
          __builtin_prefetch(WcB + ((size_t)g * H_ + d) * K2H, 0, 1);

        for (int kb = 0; kb < K2H; kb += 32) {
          // A row j = [ h_j (K<512) | h_{j+1} (K>=512) ]
          int scol, srow0, srow1;
          if (kb < H_) { scol = kb;      srow0 = M0 + lrow;     srow1 = M0 + 16 + lrow; }
          else         { scol = kb - H_; srow0 = M0 + lrow + 1; srow1 = M0 + 17 + lrow;
                         if (srow1 > L_ - 1) srow1 = L_ - 1; }
          union { v16bf v; v8bf h2[2]; } a0, a1;
          const bf16_t* ap0 = hlds + srow0 * H_ + scol + 8 * lhi;
          a0.h2[0] = *(const v8bf*)(ap0);
          a0.h2[1] = *(const v8bf*)(ap0 + 16);
          const bf16_t* ap1 = hlds + srow1 * H_ + scol + 8 * lhi;
          a1.h2[0] = *(const v8bf*)(ap1);
          a1.h2[1] = *(const v8bf*)(ap1 + 16);
#pragma unroll
          for (int g = 0; g < 5; ++g) {
            union { v16bf v; v8bf h2[2]; } bu;
            const bf16_t* bp = WcB + ((size_t)g * H_ + d) * K2H + kb + 8 * lhi;
            bu.h2[0] = *(const v8bf*)(bp);
            bu.h2[1] = *(const v8bf*)(bp + 16);
            v16bf bfrag = bu.v;
            acc[0][g] = __builtin_amdgcn_wmma_f32_16x16x32_bf16(false, a0.v, false, bfrag,
                                                                (short)0, acc[0][g], false, false);
            if (doM1)
              acc[1][g] = __builtin_amdgcn_wmma_f32_16x16x32_bf16(false, a1.v, false, bfrag,
                                                                  (short)0, acc[1][g], false, false);
          }
        }

        // fused cell epilogue on accumulator registers
        float bcI  = bc[d];
        float bcFl = bc[H_ + d];
        float bcFr = bc[2 * H_ + d];
        float bcU  = bc[3 * H_ + d];
        float bcO  = bc[4 * H_ + d];
#pragma unroll
        for (int mm = 0; mm < 2; ++mm) {
          if (mm == 1 && !doM1) break;
#pragma unroll
          for (int r = 0; r < 8; ++r) {
            int j = M0 + mm * 16 + r + 8 * lhi;
            if (j >= Lp) continue;
            float vi  = acc[mm][0][r] + bcI;
            float vfl = acc[mm][1][r] + bcFl;
            float vfr = acc[mm][2][r] + bcFr;
            float vu  = acc[mm][3][r] + bcU;
            float vo  = acc[mm][4][r] + bcO;
            float cl = cc[j * H_ + d];
            float cr = cc[(j + 1) * H_ + d];
            float cv = cl * sigm(vfl + 1.0f) + cr * sigm(vfr + 1.0f)
                     + tanhf(vu) * sigm(vi);
            float hv = sigm(vo) * tanhf(cv);
            hn[j * H_ + d] = hv;
            cn[j * H_ + d] = cv;
          }
        }
      }
      __threadfence();
      __syncthreads();

      // ---- phase 2: scores = nh @ q (8 threads per row), argmax ----
      {
        int j = tid >> 3;          // 0..63
        int p = tid & 7;           // 0..7 -> 64-element slice
        float s = 0.0f;
        if (j < Lp) {
          const float* hp = hn + j * H_ + p * 64;
          const float* qp = q + p * 64;
          for (int d2 = 0; d2 < 64; ++d2) s += hp[d2] * qp[d2];
        }
        part[tid] = s;
      }
      __syncthreads();
      if (tid < L_) {
        float s = -1e30f;
        if (tid < Lp) {
          s = 0.0f;
#pragma unroll
          for (int p = 0; p < 8; ++p) s += part[tid * 8 + p];
        }
        scf[tid] = s;
      }
      __syncthreads();
      if (tid == 0) {
        int best = 0; float bv = scf[0];
        for (int j = 1; j < Lp; ++j) if (scf[j] > bv) { bv = scf[j]; best = j; }
        *ksel = best;
      }
      __syncthreads();
      kstar = *ksel;
    }

    // ---- phase 3: select/shift (active) or plain truncation (inactive) ----
    // j <  kstar : keep old[j];  j == kstar : keep nh/nc (already in hn/cn);
    // j >  kstar : old[j+1];     !active    : old[j] for all j.
    for (int idx = tid; idx < Lp * H_; idx += NT_TREE) {
      int j = idx >> 9;
      if (!active || j < kstar) {
        hn[idx] = hc[idx];
        cn[idx] = cc[idx];
      } else if (j > kstar) {
        int src = idx + H_;
        hn[idx] = hc[src];
        cn[idx] = cc[src];
      }
    }
    __threadfence();
    __syncthreads();

    // swap double buffers
    float* t0 = hc; hc = hn; hn = t0;
    float* t1 = cc; cc = cn; cn = t1;
  }

  // ---- output: (h[:,0], c[:,0]) concatenated flat ----
  for (int d2 = tid; d2 < H_; d2 += NT_TREE) {
    out[(size_t)b * H_ + d2]                   = hc[d2];
    out[(size_t)B_ * H_ + (size_t)b * H_ + d2] = cc[d2];
  }
}

// ---------------------------------------------------------------------------
// Launcher
// ---------------------------------------------------------------------------
extern "C" void kernel_launch(void* const* d_in, const int* in_sizes, int n_in,
                              void* d_out, int out_size, void* d_ws, size_t ws_size,
                              hipStream_t stream) {
  const float* inp    = (const float*)d_in[0];   // (B, L, W)
  const int*   length = (const int*)  d_in[1];   // (B,)
  const float* Ww     = (const float*)d_in[2];   // (2H, W)
  const float* bw     = (const float*)d_in[3];   // (2H,)
  const float* Wc     = (const float*)d_in[4];   // (5H, 2H)
  const float* bc     = (const float*)d_in[5];   // (5H,)
  const float* q      = (const float*)d_in[6];   // (H,)
  float* out = (float*)d_out;                    // 2*B*H floats

  char* ws = (char*)d_ws;
  // workspace layout (bytes):
  bf16_t* inpB = (bf16_t*)(ws);                  //  4 MiB  (B*L*W bf16)
  bf16_t* WwB  = (bf16_t*)(ws + 4194304);        //  1 MiB  (2H*W bf16)
  bf16_t* WcB  = (bf16_t*)(ws + 5242880);        //  5 MiB  (5H*2H bf16)
  float*  hA   = (float*) (ws + 10485760);       //  8 MiB  (B*L*H f32)
  float*  cA   = (float*) (ws + 18874368);       //  8 MiB
  float*  hB   = (float*) (ws + 27262976);       //  8 MiB
  float*  cB   = (float*) (ws + 35651584);       //  8 MiB  -> 42 MiB total

  (void)in_sizes; (void)n_in; (void)out_size; (void)ws_size;

  k_convert<<<2048, 256, 0, stream>>>(inp, Ww, Wc, inpB, WwB, WcB);
  k_proj   <<<2048, 256, 0, stream>>>(inpB, WwB, bw, hA, cA);
  k_tree   <<<B_, NT_TREE, 0, stream>>>(WcB, bc, q, length, hA, cA, hB, cB, out);
}
